// GRU_40381282517741
// MI455X (gfx1250) — compile-verified, matrix operand order checked by reading, not run on hardware
//
#include <hip/hip_runtime.h>
#include <stdint.h>

#define B_   64
#define T_   512
#define D_   512
#define H_   1024
#define TB_  (T_ * B_)       // 32768
#define NBLK_REC 32          // persistent blocks: H_/32 columns each

typedef __attribute__((ext_vector_type(16))) __bf16 v16bf;
typedef __attribute__((ext_vector_type(8)))  float  v8f;

// ---------- helpers ----------

__device__ __forceinline__ unsigned short f2bf(float f) {
  unsigned u = __float_as_uint(f);
  unsigned r = (u + 0x7FFFu + ((u >> 16) & 1u)) >> 16;   // RNE
  return (unsigned short)r;
}

// 16-bit A/B WMMA fragment (16x32 / 32x16): lane&15 = row (M or N),
// lane>>4 selects K-half; 16 contiguous bf16 (32B) per lane.
__device__ __forceinline__ v16bf load_frag(const unsigned short* p, int stride,
                                           int row0, int k0, int lane) {
  const int r = row0 + (lane & 15);
  const int k = k0 + ((lane >> 4) << 4);
  return *(const v16bf*)(p + (size_t)r * (size_t)stride + (size_t)k);
}

__device__ __forceinline__ v8f wmma_bf16(v16bf a, v16bf b, v8f c) {
  return __builtin_amdgcn_wmma_f32_16x16x32_bf16(false, a, false, b,
                                                 (short)0, c, false, false);
}

__device__ __forceinline__ float sigmoidf_(float x) {
  return 1.0f / (1.0f + __expf(-x));
}

__device__ __forceinline__ void grid_barrier(unsigned* cnt, unsigned* gen,
                                             unsigned nb, unsigned target) {
  __threadfence();
  __syncthreads();
  if (threadIdx.x == 0) {
    unsigned prev = __hip_atomic_fetch_add(cnt, 1u, __ATOMIC_ACQ_REL,
                                           __HIP_MEMORY_SCOPE_AGENT);
    if (prev == nb - 1u) {
      __hip_atomic_store(cnt, 0u, __ATOMIC_RELAXED, __HIP_MEMORY_SCOPE_AGENT);
      __hip_atomic_fetch_add(gen, 1u, __ATOMIC_RELEASE, __HIP_MEMORY_SCOPE_AGENT);
    } else {
      while (__hip_atomic_load(gen, __ATOMIC_ACQUIRE,
                               __HIP_MEMORY_SCOPE_AGENT) < target)
        __builtin_amdgcn_s_sleep(1);
    }
  }
  __syncthreads();
  __threadfence();
}

// ---------- prep kernels ----------

__global__ void k_init(const float* __restrict__ state, float* __restrict__ hbuf,
                       unsigned short* __restrict__ hb16, unsigned* __restrict__ bar) {
  int i = blockIdx.x * blockDim.x + threadIdx.x;
  if (i < 2) bar[i] = 0u;
  if (i < B_ * H_) {
    float v = state[i];
    hbuf[i] = v;
    hb16[i] = f2bf(v);
  }
}

__global__ void k_cvt(const float* __restrict__ src, unsigned short* __restrict__ dst, int n) {
  int i = blockIdx.x * blockDim.x + threadIdx.x;
  if (i < n) dst[i] = f2bf(src[i]);
}

// src [R][C] fp32 -> dst [C][R] bf16 (so B fragments load like A fragments)
__global__ void k_transpose_cvt(const float* __restrict__ src,
                                unsigned short* __restrict__ dst, int R, int C) {
  int i = blockIdx.x * blockDim.x + threadIdx.x;
  if (i < R * C) {
    int r = i / C, c = i - r * C;
    dst[(size_t)c * R + r] = f2bf(src[i]);
  }
}

// ---------- phase 1: Xg[g][t][b][h] = inputs @ W_xg + b_g ----------
// grid (N/64, M/64, 3), 128 threads (4 waves); wave = 16-row strip, 4 N-tiles
__global__ void k_xproj(const unsigned short* __restrict__ xb,   // [32768][512] bf16
                        const unsigned short* __restrict__ wxt,  // [3][1024][512] bf16 (transposed)
                        const float* __restrict__ bz, const float* __restrict__ br,
                        const float* __restrict__ bh,
                        float* __restrict__ xg)                  // [3][512][64][1024]
{
  const int lane = threadIdx.x & 31;
  const int wave = threadIdx.x >> 5;                 // 0..3
  const int g  = blockIdx.z;
  const int m0 = blockIdx.y * 64 + wave * 16;        // row in [0,32768)
  const int n0 = blockIdx.x * 64;
  const float* bias = (g == 0) ? bz : (g == 1) ? br : bh;
  const unsigned short* w = wxt + (size_t)g * H_ * D_;

  v8f acc[4] = {v8f{}, v8f{}, v8f{}, v8f{}};

  // software pipeline the shared A fragment
  v16bf a = load_frag(xb, D_, m0, 0, lane);
  for (int kk = 0; kk < D_; kk += 32) {
    v16bf an = a;
    if (kk + 32 < D_) {
      an = load_frag(xb, D_, m0, kk + 32, lane);
      __builtin_prefetch(xb + (size_t)(m0 + (lane & 15)) * D_ + kk + 64, 0, 3);
    }
#pragma unroll
    for (int j = 0; j < 4; ++j) {
      v16bf b = load_frag(w, D_, n0 + j * 16, kk, lane);
      acc[j] = wmma_bf16(a, b, acc[j]);
    }
    a = an;
  }

  const int nl = lane & 15, mh = lane >> 4;
#pragma unroll
  for (int j = 0; j < 4; ++j) {
    int col = n0 + j * 16 + nl;
    float bv = bias[col];
#pragma unroll
    for (int i = 0; i < 8; ++i) {
      int r = m0 + mh * 8 + i;           // flat (b,t) row: r = b*T + t
      int b = r >> 9;                    // /512
      int t = r & (T_ - 1);
      xg[(((size_t)g * T_ + t) * B_ + b) * H_ + col] = acc[j][i] + bv;
    }
  }
}

// ---------- phase 2: persistent recurrent scan ----------
// 32 blocks x 256 threads (8 waves). Block owns 32 columns of H;
// wave (w>>1) -> 16-row strip of B, (w&1) -> 16-col half.
// The block's weight slices (3 gates x 32 cols x 1024 K, bf16 = 192 KB) are
// cached in LDS once and reused for all 512 timesteps (320 KB WGP LDS).
__global__ void k_recurrent(float* __restrict__ hbuf,            // [64][1024] fp32
                            unsigned short* __restrict__ hb16,   // [64][1024] bf16
                            unsigned short* __restrict__ rh,     // [64][1024] bf16 (R*H)
                            const unsigned short* __restrict__ wht, // [3][1024][1024] bf16 transposed (z,r,h)
                            const float* __restrict__ xg,        // [3][512][64][1024]
                            float* __restrict__ out,             // [1024][32768]
                            float* __restrict__ hfin,            // [64][1024]
                            unsigned* __restrict__ bar)
{
  __shared__ unsigned short wlds[3 * 32 * H_];       // 192 KB of 320 KB WGP LDS

  const int lane = threadIdx.x & 31;
  const int wave = threadIdx.x >> 5;                  // 0..7
  const int m0 = (wave >> 1) * 16;                    // 0,16,32,48
  const int nb0 = blockIdx.x * 32;
  const int nloc = (wave & 1) * 16;                   // column half within block
  unsigned* cnt = bar;
  unsigned* gen = bar + 1;
  const int nl = lane & 15, mh = lane >> 4;
  const int col = nb0 + nloc + nl;
  const int ko = (lane >> 4) << 4;                    // K-half offset
  const unsigned nb = gridDim.x;
  unsigned tgt = 0;

  // ---- stage this block's weight columns into LDS (once) ----
  {
    const int total = 3 * 32 * (H_ / 4);              // u64 chunks
    for (int i = threadIdx.x; i < total; i += blockDim.x) {
      int e = i << 2;                                 // ushort element index
      int g = e >> 15;                                // /(32*1024)
      int rem = e & 32767;
      int c = rem >> 10;
      int k = rem & (H_ - 1);
      const unsigned short* src = wht + ((size_t)g * H_ + nb0 + c) * H_ + k;
      *(uint64_t*)&wlds[e] = *(const uint64_t*)src;
    }
  }
  __syncthreads();

  // per-lane row base pointers (fragment rows fixed for whole scan)
  const unsigned short* wz_row = &wlds[(0 * 32 + nloc + nl) * H_ + ko];
  const unsigned short* wr_row = &wlds[(1 * 32 + nloc + nl) * H_ + ko];
  const unsigned short* wh_row = &wlds[(2 * 32 + nloc + nl) * H_ + ko];
  const unsigned short* a_row  = hb16 + (size_t)(m0 + nl) * H_ + ko;
  const unsigned short* r_row  = rh   + (size_t)(m0 + nl) * H_ + ko;

  for (int t = 0; t < T_; ++t) {
    // ---- Z and R GEMMs: Hc @ W_hz, Hc @ W_hr (B operands from LDS) ----
    v8f az = {}, ar = {};
    {
      v16bf a = *(const v16bf*)(a_row);
      for (int kk = 0; kk < H_; kk += 32) {
        v16bf an = a;
        if (kk + 32 < H_) an = *(const v16bf*)(a_row + kk + 32);
        v16bf fz = *(const v16bf*)(wz_row + kk);
        v16bf fr = *(const v16bf*)(wr_row + kk);
        az = wmma_bf16(a, fz, az);
        ar = wmma_bf16(a, fr, ar);
        a = an;
      }
    }

    const float* xz = xg + (size_t)t * (B_ * H_);
    const float* xr = xg + (size_t)(T_ + t) * (B_ * H_);
    const float* xh = xg + (size_t)(2 * T_ + t) * (B_ * H_);

    float zreg[8], hreg[8];
#pragma unroll
    for (int i = 0; i < 8; ++i) {
      int row = m0 + mh * 8 + i;
      float z  = sigmoidf_(az[i] + xz[(size_t)row * H_ + col]);
      float r  = sigmoidf_(ar[i] + xr[(size_t)row * H_ + col]);
      float hc = hbuf[(size_t)row * H_ + col];
      zreg[i] = z;
      hreg[i] = hc;
      rh[(size_t)row * H_ + col] = f2bf(r * hc);
    }

    ++tgt;
    grid_barrier(cnt, gen, nb, tgt);   // RH complete everywhere

    // ---- H_tilda GEMM: (R*Hc) @ W_hh ----
    v8f at = {};
    {
      v16bf a = *(const v16bf*)(r_row);
      for (int kk = 0; kk < H_; kk += 32) {
        v16bf an = a;
        if (kk + 32 < H_) an = *(const v16bf*)(r_row + kk + 32);
        v16bf fh = *(const v16bf*)(wh_row + kk);
        at = wmma_bf16(a, fh, at);
        a = an;
      }
    }

#pragma unroll
    for (int i = 0; i < 8; ++i) {
      int row = m0 + mh * 8 + i;
      float ht = tanhf(at[i] + xh[(size_t)row * H_ + col]);
      float hn = zreg[i] * hreg[i] + (1.0f - zreg[i]) * ht;
      hbuf[(size_t)row * H_ + col] = hn;
      hb16[(size_t)row * H_ + col] = f2bf(hn);
      out[(size_t)col * TB_ + (size_t)t * B_ + row] = hn;   // outputs[h][t*B+b]
      if (t == T_ - 1) hfin[(size_t)row * H_ + col] = hn;
    }

    ++tgt;
    grid_barrier(cnt, gen, nb, tgt);   // Hnew complete before next step
  }
}

// ---------- launch ----------

extern "C" void kernel_launch(void* const* d_in, const int* in_sizes, int n_in,
                              void* d_out, int out_size, void* d_ws, size_t ws_size,
                              hipStream_t stream) {
  const float* inputs = (const float*)d_in[0];
  const float* state  = (const float*)d_in[1];
  const float* W_xz = (const float*)d_in[2];
  const float* W_hz = (const float*)d_in[3];
  const float* b_z  = (const float*)d_in[4];
  const float* W_xr = (const float*)d_in[5];
  const float* W_hr = (const float*)d_in[6];
  const float* b_r  = (const float*)d_in[7];
  const float* W_xh = (const float*)d_in[8];
  const float* W_hh = (const float*)d_in[9];
  const float* b_h  = (const float*)d_in[10];

  char* ws = (char*)d_ws;
  size_t off = 0;
  unsigned*        bar  = (unsigned*)(ws + off);        off += 256;
  float*           hbuf = (float*)(ws + off);           off += (size_t)B_ * H_ * 4;
  unsigned short*  hb16 = (unsigned short*)(ws + off);  off += (size_t)B_ * H_ * 2;
  unsigned short*  rh   = (unsigned short*)(ws + off);  off += (size_t)B_ * H_ * 2;
  unsigned short*  wht  = (unsigned short*)(ws + off);  off += 3 * (size_t)H_ * H_ * 2;
  unsigned short*  wxt  = (unsigned short*)(ws + off);  off += 3 * (size_t)H_ * D_ * 2;
  unsigned short*  xb   = (unsigned short*)(ws + off);  off += (size_t)B_ * T_ * D_ * 2;
  float*           xg   = (float*)(ws + off);           off += 3 * (size_t)T_ * B_ * H_ * 4;
  (void)ws_size; (void)in_sizes; (void)n_in; (void)out_size;

  float* out  = (float*)d_out;                 // [1024][32768]
  float* hfin = out + (size_t)H_ * TB_;        // [64][1024]

  // prep
  k_init<<<(B_ * H_ + 255) / 256, 256, 0, stream>>>(state, hbuf, hb16, bar);
  {
    int n = B_ * T_ * D_;
    k_cvt<<<(n + 255) / 256, 256, 0, stream>>>(inputs, xb, n);
  }
  {
    int n = D_ * H_;
    k_transpose_cvt<<<(n + 255) / 256, 256, 0, stream>>>(W_xz, wxt + 0 * (size_t)H_ * D_, D_, H_);
    k_transpose_cvt<<<(n + 255) / 256, 256, 0, stream>>>(W_xr, wxt + 1 * (size_t)H_ * D_, D_, H_);
    k_transpose_cvt<<<(n + 255) / 256, 256, 0, stream>>>(W_xh, wxt + 2 * (size_t)H_ * D_, D_, H_);
  }
  {
    int n = H_ * H_;
    k_transpose_cvt<<<(n + 255) / 256, 256, 0, stream>>>(W_hz, wht + 0 * (size_t)H_ * H_, H_, H_);
    k_transpose_cvt<<<(n + 255) / 256, 256, 0, stream>>>(W_hr, wht + 1 * (size_t)H_ * H_, H_, H_);
    k_transpose_cvt<<<(n + 255) / 256, 256, 0, stream>>>(W_hh, wht + 2 * (size_t)H_ * H_, H_, H_);
  }

  // phase 1: input-side projections (3 big WMMA GEMMs)
  dim3 g1(H_ / 64, TB_ / 64, 3);
  k_xproj<<<g1, 128, 0, stream>>>(xb, wxt, b_z, b_r, b_h, xg);

  // phase 2: persistent recurrent scan with grid barriers + LDS weight cache
  k_recurrent<<<NBLK_REC, 256, 0, stream>>>(hbuf, hb16, rh, wht, xg, out, hfin, bar);
}
